// _EncoderLayer_54305566490658
// MI455X (gfx1250) — compile-verified, matrix operand order checked by reading, not run on hardware
//
#include <hip/hip_runtime.h>
#include <hip/hip_bf16.h>
#include <math.h>

// ---------------------------------------------------------------------------
// Transformer encoder layer for MI455X (gfx1250, wave32, WMMA f16->f32).
// B=4 S=2048 D=512 H=8 HD=64 E=4
// ---------------------------------------------------------------------------

constexpr int Bc = 4, Sc = 2048, Dc = 512, Hc = 8, HDc = 64, DEc = 2048;

typedef _Float16 v16h __attribute__((ext_vector_type(16)));
typedef _Float16 v8h  __attribute__((ext_vector_type(8)));
typedef _Float16 v4h  __attribute__((ext_vector_type(4)));
typedef float    v8f  __attribute__((ext_vector_type(8)));
typedef float    v4f  __attribute__((ext_vector_type(4)));

__device__ __forceinline__ v8f zero8() {
    v8f z;
#pragma unroll
    for (int i = 0; i < 8; ++i) z[i] = 0.0f;
    return z;
}

// Load one 16x32 f16 operand fragment from an LDS tile stored row-major
// [row][k] with `stride` halfs per row. Matches the CDNA5 16-bit A layout
// (ISA 7.12.2): lanes 0-15 -> row = lane, K = {0..7, 16..23};
// lanes 16-31 -> row = lane-16, K = {8..15, 24..31}.
// B uses the mirrored layout with N as lane index, so staging B transposed
// ([N][K]) lets this same loader produce B fragments.
__device__ __forceinline__ v16h ld_frag(const _Float16* base, int stride,
                                        int row0, int k0) {
    const int lane = threadIdx.x & 31;
    const _Float16* p = base + (row0 + (lane & 15)) * stride + k0 + ((lane >> 4) << 3);
    v8h lo = *(const v8h*)p;          // ds_load_b128
    v8h hi = *(const v8h*)(p + 16);   // ds_load_b128
    return __builtin_shufflevector(lo, hi, 0, 1, 2, 3, 4, 5, 6, 7,
                                   8, 9, 10, 11, 12, 13, 14, 15);
}

__device__ __forceinline__ float gelu_tanh(float x) {
    const float k0 = 0.7978845608028654f;  // sqrt(2/pi)
    const float k1 = 0.044715f;
    return 0.5f * x * (1.0f + tanhf(k0 * (x + k1 * x * x * x)));
}

// ---------------------------------------------------------------------------
// Generic WMMA GEMM: C[M,N] = epilogue(A[M,K] @ Bw[K,N]).
// 256 threads = 8 waves; block tile 128x128, K-step 32, double-buffered LDS
// ping-pong: next K-tile's global loads are issued before the WMMA block and
// committed to the alternate buffer after it -> one barrier per K-step and
// global latency hidden under the 8 WMMAs.
// A may be fp32 or f16 (converted to f16 in LDS); Bw is fp32 weights.
// Epilogue: +bias, optional tanh-GELU, *outScale, +residual, f32/f16 store.
// ---------------------------------------------------------------------------
template <bool AHALF, bool OUTHALF, bool GELU>
__global__ __launch_bounds__(256) void gemm_kernel(
    const void* __restrict__ Ap, const float* __restrict__ Bw,
    const float* __restrict__ bias, const float* __restrict__ resid,
    void* __restrict__ Cp, int M, int N, int K, float outScale) {
    __shared__ __align__(16) _Float16 As[2][128 * 32];
    __shared__ __align__(16) _Float16 Bs[2][128 * 32];  // transposed [n][k]

    const int tid = threadIdx.x;
    const int lane = tid & 31;
    const int wid = tid >> 5;
    const int wm = wid & 3;   // M wave coord (0..3) -> 32 rows each
    const int wn = wid >> 2;  // N wave coord (0..1) -> 64 cols each
    const int bm = blockIdx.y * 128;
    const int bnb = blockIdx.x * 128;

    const float* Af = (const float*)Ap;
    const _Float16* Ah = (const _Float16*)Ap;

    // kb-invariant staging coordinates (4 chunks of 4 elements per thread)
    int amRow[4], akCol[4], bkRow[4], bnCol[4];
#pragma unroll
    for (int j = 0; j < 4; ++j) {
        const int elem = (tid + j * 256) * 4;
        amRow[j] = elem >> 5;   // A: 128x32, row-major, k fast
        akCol[j] = elem & 31;
        bkRow[j] = elem >> 7;   // B: 32x128, row-major, n fast
        bnCol[j] = elem & 127;
    }

    v4h aH[4];
    v4f aF[4];
    v4f bF[4];

    auto load_gmem = [&](int kb) {
#pragma unroll
        for (int j = 0; j < 4; ++j) {
            if (AHALF)
                aH[j] = *(const v4h*)(Ah + (size_t)(bm + amRow[j]) * K + kb + akCol[j]);
            else
                aF[j] = *(const v4f*)(Af + (size_t)(bm + amRow[j]) * K + kb + akCol[j]);
            bF[j] = *(const v4f*)(Bw + (size_t)(kb + bkRow[j]) * N + bnb + bnCol[j]);
        }
    };
    auto store_lds = [&](int buf) {
#pragma unroll
        for (int j = 0; j < 4; ++j) {
            v4h hv;
            if (AHALF) {
                hv = aH[j];
            } else {
                hv.x = (_Float16)aF[j].x;
                hv.y = (_Float16)aF[j].y;
                hv.z = (_Float16)aF[j].z;
                hv.w = (_Float16)aF[j].w;
            }
            *(v4h*)(As[buf] + amRow[j] * 32 + akCol[j]) = hv;
            Bs[buf][(bnCol[j] + 0) * 32 + bkRow[j]] = (_Float16)bF[j].x;
            Bs[buf][(bnCol[j] + 1) * 32 + bkRow[j]] = (_Float16)bF[j].y;
            Bs[buf][(bnCol[j] + 2) * 32 + bkRow[j]] = (_Float16)bF[j].z;
            Bs[buf][(bnCol[j] + 3) * 32 + bkRow[j]] = (_Float16)bF[j].w;
        }
    };

    v8f acc[2][4];
#pragma unroll
    for (int i = 0; i < 2; ++i)
#pragma unroll
        for (int j = 0; j < 4; ++j) acc[i][j] = zero8();

    // prologue: stage first K-tile
    load_gmem(0);
    store_lds(0);
    __syncthreads();

    int cur = 0;
    for (int kb = 0; kb < K; kb += 32) {
        const bool has_next = (kb + 32) < K;
        if (has_next) load_gmem(kb + 32);  // issue early, consumed after WMMAs

        v16h af[2], bf[4];
#pragma unroll
        for (int mi = 0; mi < 2; ++mi)
            af[mi] = ld_frag(As[cur], 32, wm * 32 + mi * 16, 0);
#pragma unroll
        for (int ni = 0; ni < 4; ++ni)
            bf[ni] = ld_frag(Bs[cur], 32, wn * 64 + ni * 16, 0);

#pragma unroll
        for (int mi = 0; mi < 2; ++mi)
#pragma unroll
            for (int ni = 0; ni < 4; ++ni)
                acc[mi][ni] = __builtin_amdgcn_wmma_f32_16x16x32_f16(
                    false, af[mi], false, bf[ni], (short)0, acc[mi][ni], false, false);

        if (has_next) store_lds(cur ^ 1);
        __syncthreads();
        cur ^= 1;
    }

    // ---- epilogue ----
    float* Cf = (float*)Cp;
    _Float16* Ch = (_Float16*)Cp;
#pragma unroll
    for (int mi = 0; mi < 2; ++mi) {
#pragma unroll
        for (int ni = 0; ni < 4; ++ni) {
            const int n = bnb + wn * 64 + ni * 16 + (lane & 15);
            const int m0 = bm + wm * 32 + mi * 16 + ((lane >> 4) << 3);
            const float bv = bias ? bias[n] : 0.0f;
#pragma unroll
            for (int r = 0; r < 8; ++r) {
                float v = acc[mi][ni][r] + bv;
                if (GELU) v = gelu_tanh(v);
                v *= outScale;
                const size_t idx = (size_t)(m0 + r) * N + n;
                if (resid) v += resid[idx];
                if (OUTHALF)
                    Ch[idx] = (_Float16)v;
                else
                    Cf[idx] = v;
            }
        }
    }
}

// ---------------------------------------------------------------------------
// Fused flash attention: o = softmax(q k^T) v  per (b,h).
// q/k/v/o: f16 [B, S, H*HD] (head h occupies columns h*64..h*64+63).
// Block: 128 threads = 4 waves; each wave = 16 query rows, full HD=64.
// K tile goes global->LDS via GLOBAL_LOAD_ASYNC_TO_LDS_B128 (ASYNCcnt, no
// VGPR round-trip); V needs a transpose so it goes through VGPRs.
// Scale D^-0.25 was folded into q and k at projection time.
// ---------------------------------------------------------------------------
__global__ __launch_bounds__(128) void flash_attn_kernel(
    const _Float16* __restrict__ q, const _Float16* __restrict__ k,
    const _Float16* __restrict__ v, _Float16* __restrict__ o) {
    __shared__ __align__(16) _Float16 Qs[64 * 64];      // [qrow][hd]
    __shared__ __align__(16) _Float16 Ks[64 * 64];      // [key][hd] == B-layout for q@k^T
    __shared__ __align__(16) _Float16 Vt[64 * 64];      // [hd][key] == B-layout for p@v
    __shared__ __align__(16) _Float16 Ps[4][16 * 64];   // per-wave P (C->A relayout)

    const int tid = threadIdx.x;
    const int lane = tid & 31;
    const int wid = tid >> 5;
    const int q0 = blockIdx.x * 64;
    const int bh = blockIdx.y;
    const int b = bh >> 3, h = bh & 7;

    const size_t headoff = (size_t)b * Sc * Dc + (size_t)h * HDc;
    const _Float16* qg = q + headoff + (size_t)q0 * Dc;

    // stage Q (64x64)
#pragma unroll
    for (int j = 0; j < 8; ++j) {
        const int c = tid + j * 128;
        const int row = c >> 4;
        const int col = (c & 15) * 4;
        *(v4h*)(Qs + row * 64 + col) = *(const v4h*)(qg + (size_t)row * Dc + col);
    }
    __syncthreads();

    const v16h aq0 = ld_frag(Qs, 64, wid * 16, 0);
    const v16h aq1 = ld_frag(Qs, 64, wid * 16, 32);

    float mrun[8], lrun[8];
    v8f oacc[4];
#pragma unroll
    for (int r = 0; r < 8; ++r) { mrun[r] = -1e30f; lrun[r] = 0.0f; }
#pragma unroll
    for (int i = 0; i < 4; ++i) oacc[i] = zero8();

    for (int kb = 0; kb < Sc; kb += 64) {
        const _Float16* kg = k + headoff + (size_t)kb * Dc;
        const _Float16* vg = v + headoff + (size_t)kb * Dc;

        // K tile: async global->LDS copy, 16B per lane per issue.
        // LDS dest address = low 32 bits of the generic pointer (LDS aperture
        // maps addr[31:0] to the LDS offset).
#pragma unroll
        for (int j = 0; j < 4; ++j) {
            const int c = tid + j * 128;
            const int row = c >> 3;          // key index within tile
            const int col = (c & 7) * 8;     // hd, 8 halfs = 16B
            const unsigned ldsoff = (unsigned)(uintptr_t)(Ks + row * 64 + col);
            const unsigned long long ga =
                (unsigned long long)(uintptr_t)(kg + (size_t)row * Dc + col);
            asm volatile("global_load_async_to_lds_b128 %0, %1, off"
                         :: "v"(ldsoff), "v"(ga) : "memory");
        }
        // V tile: load + transpose into Vt[hd][key]
#pragma unroll
        for (int j = 0; j < 8; ++j) {
            const int c = tid + j * 128;
            const int row = c >> 4;
            const int col = (c & 15) * 4;
            v4h vv = *(const v4h*)(vg + (size_t)row * Dc + col);
            Vt[(col + 0) * 64 + row] = vv.x;
            Vt[(col + 1) * 64 + row] = vv.y;
            Vt[(col + 2) * 64 + row] = vv.z;
            Vt[(col + 3) * 64 + row] = vv.w;
        }
        asm volatile("s_wait_asynccnt 0" ::: "memory");
        __syncthreads();

        // S = Q @ K^T : 4 tiles of 16x16, each accumulated over HD=64 (2 WMMAs)
        v8f sc[4];
#pragma unroll
        for (int nt = 0; nt < 4; ++nt) {
            v16h bk0 = ld_frag(Ks, 64, nt * 16, 0);
            v16h bk1 = ld_frag(Ks, 64, nt * 16, 32);
            v8f c = zero8();
            c = __builtin_amdgcn_wmma_f32_16x16x32_f16(false, aq0, false, bk0,
                                                       (short)0, c, false, false);
            c = __builtin_amdgcn_wmma_f32_16x16x32_f16(false, aq1, false, bk1,
                                                       (short)0, c, false, false);
            sc[nt] = c;
        }

        // online softmax; C-layout: lanes 0-15 hold rows 0-7 (elem r -> row r),
        // lanes 16-31 hold rows 8-15. Row reductions = xor-shuffles within the
        // 16-lane half (masks 1,2,4,8 never cross halves on wave32).
        _Float16* P = &Ps[wid][0];
#pragma unroll
        for (int r = 0; r < 8; ++r) {
            float mx = fmaxf(fmaxf(sc[0][r], sc[1][r]), fmaxf(sc[2][r], sc[3][r]));
#pragma unroll
            for (int s = 1; s < 16; s <<= 1) mx = fmaxf(mx, __shfl_xor(mx, s, 32));
            const float mnew = fmaxf(mrun[r], mx);
            const float alpha = __expf(mrun[r] - mnew);
            const int prow = ((lane >> 4) << 3) + r;
            float rowsum = 0.0f;
#pragma unroll
            for (int nt = 0; nt < 4; ++nt) {
                const float p = __expf(sc[nt][r] - mnew);
                rowsum += p;
                P[prow * 64 + nt * 16 + (lane & 15)] = (_Float16)p;
            }
#pragma unroll
            for (int s = 1; s < 16; s <<= 1) rowsum += __shfl_xor(rowsum, s, 32);
            lrun[r] = lrun[r] * alpha + rowsum;
            mrun[r] = mnew;
#pragma unroll
            for (int nt = 0; nt < 4; ++nt) oacc[nt][r] *= alpha;
        }
        // wave-local fence: P stores -> P fragment reloads (DS counter only)
        asm volatile("s_wait_dscnt 0" ::: "memory");

        // O += P @ V
        const v16h ap0 = ld_frag(P, 64, 0, 0);
        const v16h ap1 = ld_frag(P, 64, 0, 32);
#pragma unroll
        for (int nt = 0; nt < 4; ++nt) {
            v16h bv0 = ld_frag(Vt, 64, nt * 16, 0);
            v16h bv1 = ld_frag(Vt, 64, nt * 16, 32);
            oacc[nt] = __builtin_amdgcn_wmma_f32_16x16x32_f16(
                false, ap0, false, bv0, (short)0, oacc[nt], false, false);
            oacc[nt] = __builtin_amdgcn_wmma_f32_16x16x32_f16(
                false, ap1, false, bv1, (short)0, oacc[nt], false, false);
        }
        __syncthreads();  // before next iteration overwrites Ks/Vt
    }

    // writeout: o[b, q0+row, h*64+col] = oacc / l
    _Float16* og = o + headoff + (size_t)q0 * Dc;
#pragma unroll
    for (int r = 0; r < 8; ++r) {
        const int row = wid * 16 + ((lane >> 4) << 3) + r;
        const float invl = 1.0f / lrun[r];
#pragma unroll
        for (int nt = 0; nt < 4; ++nt)
            og[(size_t)row * Dc + nt * 16 + (lane & 15)] =
                (_Float16)(oacc[nt][r] * invl);
    }
}

// ---------------------------------------------------------------------------
extern "C" void kernel_launch(void* const* d_in, const int* in_sizes, int n_in,
                              void* d_out, int out_size, void* d_ws, size_t ws_size,
                              hipStream_t stream) {
    (void)in_sizes; (void)n_in; (void)out_size; (void)ws_size;
    const float* x  = (const float*)d_in[0];
    const float* Wq = (const float*)d_in[1];
    const float* Wk = (const float*)d_in[2];
    const float* Wv = (const float*)d_in[3];
    const float* Wo = (const float*)d_in[4];
    const float* W1 = (const float*)d_in[5];
    const float* b1 = (const float*)d_in[6];
    const float* W2 = (const float*)d_in[7];
    const float* b2 = (const float*)d_in[8];
    float* out = (float*)d_out;

    const size_t NE = (size_t)Bc * Sc * Dc;  // 4,194,304
    _Float16* qh = (_Float16*)d_ws;
    _Float16* kh = qh + NE;
    _Float16* vh = kh + NE;
    _Float16* oh = vh + NE;
    float* x1 = (float*)(oh + NE);
    _Float16* hh = (_Float16*)(x1 + NE);     // B*S*DE halfs

    const int M = Bc * Sc;                    // 8192
    const float scale = powf((float)Dc, -0.25f);  // applied to q AND k

    dim3 blk(256);
    // q/k/v projections (scale folded into q,k), f16 outputs
    gemm_kernel<false, true, false><<<dim3(Dc / 128, M / 128), blk, 0, stream>>>(
        x, Wq, nullptr, nullptr, qh, M, Dc, Dc, scale);
    gemm_kernel<false, true, false><<<dim3(Dc / 128, M / 128), blk, 0, stream>>>(
        x, Wk, nullptr, nullptr, kh, M, Dc, Dc, scale);
    gemm_kernel<false, true, false><<<dim3(Dc / 128, M / 128), blk, 0, stream>>>(
        x, Wv, nullptr, nullptr, vh, M, Dc, Dc, 1.0f);

    // fused attention
    flash_attn_kernel<<<dim3(Sc / 64, Bc * Hc), 128, 0, stream>>>(qh, kh, vh, oh);

    // x1 = o @ Wo + x
    gemm_kernel<true, false, false><<<dim3(Dc / 128, M / 128), blk, 0, stream>>>(
        oh, Wo, nullptr, x, x1, M, Dc, Dc, 1.0f);

    // h = gelu(x1 @ W1 + b1)  (f16)
    gemm_kernel<false, true, true><<<dim3(DEc / 128, M / 128), blk, 0, stream>>>(
        x1, W1, b1, nullptr, hh, M, DEc, Dc, 1.0f);

    // out = h @ W2 + b2 + x1
    gemm_kernel<true, false, false><<<dim3(Dc / 128, M / 128), blk, 0, stream>>>(
        hh, W2, b2, x1, out, M, Dc, DEc, 1.0f);
}
